// TransConvLayer_51591147160266
// MI455X (gfx1250) — matrix operand order repaired and checked by reading, not compiled, not run on hardware
//
#include <hip/hip_runtime.h>
#include <hip/hip_bf16.h>

// ---------------------------------------------------------------------------
// TransConv (relational linear attention) for MI455X / gfx1250, wave32 WMMA.
//
//   Q_r/K_r/V_r = x @ W*_r                  (WMMA bf16, packed operands)
//   kvs_r       = K_r^T diag(deg_in_r) V_r  (packed split-K WMMA + f32 atomics)
//   ks_sum_r    = sum_c deg_in_r[c] K_r[c,:]
//   S_r[n]      = sum_{e:row=n,type=r} V_r[col_e]   (edge scatter, f32 atomics)
//   norm_r(n)   = Q_r[n].ks_sum_r + Nr
//   x_out[n]   += (deg_out_r(n)*(Q_r[n]@kvs_r) + Nr*S_r[n]) / norm_r(n)
//   out         = x_out @ Whp               (WMMA bf16)
//
// All heavy GEMMs consume bf16 operands pre-packed in WMMA fragment-major
// layout: fragment = 32 lanes x 32 contiguous bytes (2x global_load_b128).
// ---------------------------------------------------------------------------

#define DD 256   // feature dim (in == out, H=1)

typedef __attribute__((ext_vector_type(16))) __bf16          v16bf;
typedef __attribute__((ext_vector_type(16))) unsigned short  v16us;
typedef __attribute__((ext_vector_type(8)))  float           v8f;
typedef unsigned short u16;

__device__ __forceinline__ u16 f2bf(float f) {
  union { float f; unsigned u; } x; x.f = f;
  unsigned u = x.u + 0x7FFFu + ((x.u >> 16) & 1u);   // round-to-nearest-even
  return (u16)(u >> 16);
}
__device__ __forceinline__ v16bf as_bf(v16us u) {
  union { v16us u; v16bf b; } x; x.u = u; return x.b;
}
__device__ __forceinline__ v8f wmma_bf16(v16us a, v16us b, v8f c) {
  return __builtin_amdgcn_wmma_f32_16x16x32_bf16(false, as_bf(a), false, as_bf(b),
                                                 (short)0, c, false, false);
}

// ---------------------------------------------------------------------------
// Pack A[M x K] (row-major f32, M%16==0, lda=K) -> bf16 A-fragments.
// Fragment f = (mb*kblks + kb); lane owns 16 bf16 at Apk[(f*32+lane)*16].
// A 16x32 layout: lane r_=lane&15 is row; pairs at K = {0,2,4,6,16,18,20,22}+8*half.
// ---------------------------------------------------------------------------
__global__ __launch_bounds__(256)
void pack_a_kernel(const float* __restrict__ A, u16* __restrict__ Apk,
                   int mblks, int kblks, int lda) {
  const int wave = threadIdx.x >> 5, lane = threadIdx.x & 31;
  const int f = blockIdx.x * 8 + wave;
  if (f >= mblks * kblks) return;
  const int mb = f / kblks, kb = f % kblks;
  const int r_ = lane & 15, half = lane >> 4;
  const float* ap = A + (size_t)(mb * 16 + r_) * lda + kb * 32 + half * 8;
  v16us au;
#pragma unroll
  for (int p = 0; p < 8; ++p) {
    const int ko = (p < 4) ? (2 * p) : (8 + 2 * p);
    au[2 * p]     = f2bf(ap[ko]);
    au[2 * p + 1] = f2bf(ap[ko + 1]);
  }
  *(v16us*)(Apk + ((size_t)f * 32 + lane) * 16) = au;
}

// Pack A^T with row weights: target [256 x kblks*32], elem(m,c) = w[c]*Krm[c*DD+m].
__global__ __launch_bounds__(256)
void pack_aT_kernel(const float* __restrict__ Krm, const float* __restrict__ w,
                    u16* __restrict__ Apk, int Nrows, int kblks) {
  const int wave = threadIdx.x >> 5, lane = threadIdx.x & 31;
  const int f = blockIdx.x * 8 + wave;
  if (f >= 16 * kblks) return;
  const int mb = f / kblks, kb = f % kblks;
  const int r_ = lane & 15, half = lane >> 4;
  v16us au;
#pragma unroll
  for (int p = 0; p < 8; ++p) {
    const int ko = ((p < 4) ? (2 * p) : (8 + 2 * p)) + half * 8;
    const int c = kb * 32 + ko;
    const float a0 = (c     < Nrows) ? w[c]     * Krm[(size_t)c * DD + mb * 16 + r_]       : 0.f;
    const float a1 = (c + 1 < Nrows) ? w[c + 1] * Krm[(size_t)(c + 1) * DD + mb * 16 + r_] : 0.f;
    au[2 * p]     = f2bf(a0);
    au[2 * p + 1] = f2bf(a1);
  }
  *(v16us*)(Apk + ((size_t)f * 32 + lane) * 16) = au;
}

// Pack B[Krows x Nn] (row-major f32) -> bf16 B-fragments, frag f = (kb*nblks + nb).
// B 32x16 layout: lane col = nb*16 + r_; pairs at K = kb*32 + {0,2,4,6,16,18,20,22}+8*half.
__global__ __launch_bounds__(256)
void pack_b_kernel(const float* __restrict__ B, u16* __restrict__ Bpk,
                   int Krows, int kblks, int nblks, int Nn) {
  const int wave = threadIdx.x >> 5, lane = threadIdx.x & 31;
  const int f = blockIdx.x * 8 + wave;
  if (f >= kblks * nblks) return;
  const int kb = f / nblks, nb = f % nblks;
  const int r_ = lane & 15, half = lane >> 4;
  v16us bu;
#pragma unroll
  for (int v = 0; v < 8; ++v) {
    const int k = kb * 32 + ((v < 4) ? (2 * v) : (8 + 2 * v)) + half * 8;
    const float b0 = (k     < Krows) ? B[(size_t)k * Nn + nb * 16 + r_]       : 0.f;
    const float b1 = (k + 1 < Krows) ? B[(size_t)(k + 1) * Nn + nb * 16 + r_] : 0.f;
    bu[2 * v]     = f2bf(b0);
    bu[2 * v + 1] = f2bf(b1);
  }
  *(v16us*)(Bpk + ((size_t)f * 32 + lane) * 16) = bu;
}

// ---------------------------------------------------------------------------
// C[mblks*16 x nblks*16] = Apk x Bpk. One wave computes 16x64 (4 n-tiles):
// per K-block: 1 A-frag + 4 B-frags (10x global_load_b128) -> 4 WMMAs.
// ---------------------------------------------------------------------------
__global__ __launch_bounds__(256)
void gemm_packed_wmma(const u16* __restrict__ Apk, const u16* __restrict__ Bpk,
                      float* __restrict__ C, int mblks, int nblks, int kblks) {
  const int wave = threadIdx.x >> 5, lane = threadIdx.x & 31;
  const int ngroups = nblks >> 2;
  const int task = blockIdx.x * 8 + wave;
  if (task >= mblks * ngroups) return;            // wave-uniform
  const int mb = task / ngroups, ng = task % ngroups;
  const int r_ = lane & 15, half = lane >> 4;

  const v16us* Af = (const v16us*)Apk + ((size_t)mb * kblks) * 32 + lane;
  const v16us* Bf = (const v16us*)Bpk + ((size_t)(ng * 4)) * 32 + lane;

  v8f acc0 = {}, acc1 = {}, acc2 = {}, acc3 = {};
  for (int kb = 0; kb < kblks; ++kb) {
    const v16us a  = Af[(size_t)kb * 32];
    const v16us* bb = Bf + (size_t)kb * nblks * 32;
    if (kb + 1 < kblks) {                         // global_prefetch_b8 next K-block
      __builtin_prefetch(Af + (size_t)(kb + 1) * 32, 0, 1);
      __builtin_prefetch(Bf + (size_t)(kb + 1) * nblks * 32, 0, 1);
    }
    const v16us b0 = bb[0], b1 = bb[32], b2 = bb[64], b3 = bb[96];
    acc0 = wmma_bf16(a, b0, acc0);
    acc1 = wmma_bf16(a, b1, acc1);
    acc2 = wmma_bf16(a, b2, acc2);
    acc3 = wmma_bf16(a, b3, acc3);
  }
  const int Nn = nblks * 16;
  float* Cp = C + (size_t)(mb * 16 + half * 8) * Nn + ng * 64 + r_;
#pragma unroll
  for (int i = 0; i < 8; ++i) {
    float* row = Cp + (size_t)i * Nn;
    row[0] = acc0[i]; row[16] = acc1[i]; row[32] = acc2[i]; row[48] = acc3[i];
  }
}

// kvs[256x256] += Apk(16 x kblksTot frags) x Bpk(kblksTot x 16 frags), split-K.
// 8 blocks (=64 wave-tasks of 16x64) per chunk; f32 atomics combine partials.
__global__ __launch_bounds__(256)
void kvs_packed_wmma(const u16* __restrict__ Apk, const u16* __restrict__ Bpk,
                     float* __restrict__ kvs, int kblksTot, int kbPerChunk) {
  const int wave = threadIdx.x >> 5, lane = threadIdx.x & 31;
  const int chunkId = blockIdx.x >> 3;
  const int task = ((blockIdx.x & 7) << 3) + wave;  // 0..63
  const int mb = task >> 2, ng = task & 3;
  const int r_ = lane & 15, half = lane >> 4;
  const int kb0 = chunkId * kbPerChunk;
  int kb1 = kb0 + kbPerChunk; if (kb1 > kblksTot) kb1 = kblksTot;

  const v16us* Af = (const v16us*)Apk + ((size_t)mb * kblksTot) * 32 + lane;
  const v16us* Bf = (const v16us*)Bpk + ((size_t)(ng * 4)) * 32 + lane;

  v8f acc0 = {}, acc1 = {}, acc2 = {}, acc3 = {};
  for (int kb = kb0; kb < kb1; ++kb) {
    const v16us a  = Af[(size_t)kb * 32];
    const v16us* bb = Bf + (size_t)kb * 16 * 32;   // nblks = 16
    const v16us b0 = bb[0], b1 = bb[32], b2 = bb[64], b3 = bb[96];
    acc0 = wmma_bf16(a, b0, acc0);
    acc1 = wmma_bf16(a, b1, acc1);
    acc2 = wmma_bf16(a, b2, acc2);
    acc3 = wmma_bf16(a, b3, acc3);
  }
  float* base = kvs + (size_t)(mb * 16 + half * 8) * DD + ng * 64 + r_;
#pragma unroll
  for (int i = 0; i < 8; ++i) {
    float* row = base + (size_t)i * DD;
    atomicAdd(row +  0, acc0[i]);
    atomicAdd(row + 16, acc1[i]);
    atomicAdd(row + 32, acc2[i]);
    atomicAdd(row + 48, acc3[i]);
  }
}

// deg_in[r*N+col] += 1, deg_out[r*N+row] += 1 over all edges
__global__ void count_kernel(const int* __restrict__ ei, const int* __restrict__ et,
                             float* __restrict__ deg_in, float* __restrict__ deg_out,
                             int E, int N) {
  const int e = blockIdx.x * 256 + threadIdx.x;
  if (e >= E) return;
  const int r = et[e];
  atomicAdd(&deg_out[(size_t)r * N + ei[e]],     1.0f);
  atomicAdd(&deg_in [(size_t)r * N + ei[E + e]], 1.0f);
}

// Nr[r] = sum_c deg_in[r][c]
__global__ void nr_kernel(const float* __restrict__ deg_in, float* __restrict__ Nr, int N) {
  __shared__ float s[256];
  const int r = blockIdx.x;
  float acc = 0.f;
  for (int i = threadIdx.x; i < N; i += 256) acc += deg_in[(size_t)r * N + i];
  s[threadIdx.x] = acc; __syncthreads();
  for (int st = 128; st > 0; st >>= 1) {
    if (threadIdx.x < st) s[threadIdx.x] += s[threadIdx.x + st];
    __syncthreads();
  }
  if (threadIdx.x == 0) Nr[r] = s[0];
}

// ks[d] += sum_c w[c]*Km[c,d] (chunked, atomics)
__global__ void kssum_kernel(const float* __restrict__ Km, const float* __restrict__ w,
                             float* __restrict__ ks, int Nnodes) {
  const int d = threadIdx.x;
  const int c0 = blockIdx.x * 256;
  float s = 0.f;
  for (int i = 0; i < 256; ++i) {
    const int c = c0 + i;
    if (c < Nnodes) s += w[c] * Km[(size_t)c * DD + d];
  }
  atomicAdd(&ks[d], s);
}

// S[row_e, :] += V[col_e, :] for edges of relation rsel (one block per edge)
__global__ void scatter_kernel(const int* __restrict__ ei, const int* __restrict__ et,
                               const float* __restrict__ V, float* __restrict__ S,
                               int E, int rsel) {
  const int e = blockIdx.x;
  if (et[e] != rsel) return;                      // uniform per block
  const int row = ei[e], col = ei[E + e];
  atomicAdd(&S[(size_t)row * DD + threadIdx.x], V[(size_t)col * DD + threadIdx.x]);
}

// x_out[n,d] += (deg_out[n]*P[n,d] + Nr*S[n,d]) / (Q[n].ks + Nr)
__global__ void finalize_kernel(const float* __restrict__ Q, const float* __restrict__ P,
                                const float* __restrict__ S, const float* __restrict__ ks,
                                const float* __restrict__ degout, const float* __restrict__ NrP,
                                float* __restrict__ xout) {
  __shared__ float s[256];
  const int n = blockIdx.x, d = threadIdx.x;
  const float Nr = NrP[0];
  s[d] = Q[(size_t)n * DD + d] * ks[d]; __syncthreads();
  for (int st = 128; st > 0; st >>= 1) {
    if (d < st) s[d] += s[d + st];
    __syncthreads();
  }
  const float norm = s[0] + Nr;
  const float val = (degout[n] * P[(size_t)n * DD + d] + Nr * S[(size_t)n * DD + d]) / norm;
  xout[(size_t)n * DD + d] += val;
}

extern "C" void kernel_launch(void* const* d_in, const int* in_sizes, int n_in,
                              void* d_out, int out_size, void* d_ws, size_t ws_size,
                              hipStream_t stream) {
  const float* x   = (const float*)d_in[0];
  const float* Wq  = (const float*)d_in[1];
  const float* Wk  = (const float*)d_in[2];
  const float* Wv  = (const float*)d_in[3];
  const float* Whp = (const float*)d_in[4];
  const int*   ei  = (const int*)d_in[5];
  const int*   et  = (const int*)d_in[6];

  const int N = in_sizes[0] / DD;           // 50000 (divisible by 16)
  const int E = in_sizes[6];                // 300000
  const int R = in_sizes[1] / (DD * DD);    // 3

  const int mblks   = N / 16;               // 3125
  const int kblksD  = DD / 32;              // 8
  const int nblksD  = DD / 16;              // 16
  const int kblksN  = (N + 31) / 32;        // 1563 (K = nodes, zero-padded)

  // ---- workspace layout ----
  float* ws = (float*)d_ws;
  const size_t nd = (size_t)N * DD;
  float* Q       = ws;                        // nd f32
  float* Kb      = Q  + nd;                   // nd f32  (reused as P)
  float* Vb      = Kb + nd;                   // nd f32
  float* S       = Vb + nd;                   // nd f32
  float* Xo      = S  + nd;                   // nd f32
  float* deg_in  = Xo + nd;                   // R*N
  float* deg_out = deg_in + (size_t)R * N;    // R*N
  float* Nr      = deg_out + (size_t)R * N;   // 16 (pad)
  float* kvs     = Nr + 16;                   // DD*DD
  float* ks      = kvs + DD * DD;             // DD
  float* P       = Kb;                        // alias: K_r dead after kvs/ks_sum

  u16* pk     = (u16*)(ks + DD);
  u16* xApk   = pk;                                   // nd bf16 (x as A)
  u16* QApk   = xApk + nd;                            // nd bf16 (Q as A; Xo reuses)
  u16* KTApk  = QApk + nd;                            // 16*kblksN*512
  u16* VBpk   = KTApk + (size_t)16 * kblksN * 512;    // kblksN*16*512
  u16* WBpk   = VBpk  + (size_t)kblksN * 16 * 512;    // 10 * 64K (3q,3k,3v,whp)
  u16* kvsBpk = WBpk + (size_t)10 * DD * DD;          // 64K

  // ---- zero accumulators ----
  hipMemsetAsync(deg_in, 0, ((size_t)2 * R * N + 16) * sizeof(float), stream);
  hipMemsetAsync(Xo, 0, nd * sizeof(float), stream);

  // ---- degree counts + Nr ----
  count_kernel<<<(E + 255) / 256, 256, 0, stream>>>(ei, et, deg_in, deg_out, E, N);
  nr_kernel<<<R, 256, 0, stream>>>(deg_in, Nr, N);

  // ---- one-time packs: x as A, all weights as B ----
  const int aFrags  = mblks * kblksD;                 // 25000
  pack_a_kernel<<<(aFrags + 7) / 8, 256, 0, stream>>>(x, xApk, mblks, kblksD, DD);
  const int wFrags = kblksD * nblksD;                 // 128
  for (int i = 0; i < 3 * R + 1; ++i) {
    const float* Wsrc = (i < R)         ? Wq  + (size_t)i * DD * DD
                      : (i < 2 * R)     ? Wk  + (size_t)(i - R) * DD * DD
                      : (i < 3 * R)     ? Wv  + (size_t)(i - 2 * R) * DD * DD
                                        : Whp;
    pack_b_kernel<<<(wFrags + 7) / 8, 256, 0, stream>>>(Wsrc, WBpk + (size_t)i * DD * DD,
                                                        DD, kblksD, nblksD, DD);
  }

  const int gTasks  = mblks * (nblksD / 4);           // 12500
  const int gBlocks = (gTasks + 7) / 8;               // 1563
  const int kbPerChunk = 25;                          // 63 chunks over 1563 k-blocks
  const int nChunks = (kblksN + kbPerChunk - 1) / kbPerChunk;
  const int tFrags  = 16 * kblksN;                    // 25008

  for (int r = 0; r < R; ++r) {
    const float* win = deg_in  + (size_t)r * N;
    const float* wot = deg_out + (size_t)r * N;

    // Q_r / K_r / V_r = x @ W*_r
    gemm_packed_wmma<<<gBlocks, 256, 0, stream>>>(xApk, WBpk + (size_t)r * DD * DD,           Q,  mblks, nblksD, kblksD);
    gemm_packed_wmma<<<gBlocks, 256, 0, stream>>>(xApk, WBpk + (size_t)(R + r) * DD * DD,     Kb, mblks, nblksD, kblksD);
    gemm_packed_wmma<<<gBlocks, 256, 0, stream>>>(xApk, WBpk + (size_t)(2 * R + r) * DD * DD, Vb, mblks, nblksD, kblksD);

    // pack Q as A, (w*K)^T as A, V as B
    pack_a_kernel <<<(aFrags + 7) / 8, 256, 0, stream>>>(Q, QApk, mblks, kblksD, DD);
    pack_aT_kernel<<<(tFrags + 7) / 8, 256, 0, stream>>>(Kb, win, KTApk, N, kblksN);
    pack_b_kernel <<<(tFrags + 7) / 8, 256, 0, stream>>>(Vb, VBpk, N, kblksN, nblksD, DD);

    // kvs_r, ks_sum_r
    hipMemsetAsync(kvs, 0, (size_t)(DD * DD + DD) * sizeof(float), stream);
    kvs_packed_wmma<<<8 * nChunks, 256, 0, stream>>>(KTApk, VBpk, kvs, kblksN, kbPerChunk);
    kssum_kernel<<<(N + 255) / 256, 256, 0, stream>>>(Kb, win, ks, N);

    // S_r edge scatter
    hipMemsetAsync(S, 0, nd * sizeof(float), stream);
    scatter_kernel<<<E, 256, 0, stream>>>(ei, et, Vb, S, E, r);

    // P_r = Q_r @ kvs_r
    pack_b_kernel<<<(wFrags + 7) / 8, 256, 0, stream>>>(kvs, kvsBpk, DD, kblksD, nblksD, DD);
    gemm_packed_wmma<<<gBlocks, 256, 0, stream>>>(QApk, kvsBpk, P, mblks, nblksD, kblksD);

    // accumulate into x_out
    finalize_kernel<<<N, 256, 0, stream>>>(Q, P, S, ks, wot, Nr + r, Xo);
  }

  // out = x_out @ Whp   (reuse QApk storage for packed Xo)
  pack_a_kernel<<<(aFrags + 7) / 8, 256, 0, stream>>>(Xo, QApk, mblks, kblksD, DD);
  gemm_packed_wmma<<<gBlocks, 256, 0, stream>>>(QApk, WBpk + (size_t)3 * R * DD * DD,
                                                (float*)d_out, mblks, nblksD, kblksD);
}
